// CrossTokenSelectorSaliency_13073880449114
// MI455X (gfx1250) — matrix operand order repaired and checked by reading, not compiled, not run on hardware
//
#include <hip/hip_runtime.h>
#include <hip/hip_bf16.h>

// ---------------------------------------------------------------------------
// CrossTokenSelectorSaliency for MI455X (gfx1250, wave32, WMMA)
// ---------------------------------------------------------------------------

typedef __bf16 bf16_t;
typedef __attribute__((ext_vector_type(16))) __bf16 v16bf;
typedef __attribute__((ext_vector_type(8)))  __bf16 v8bf;
typedef __attribute__((ext_vector_type(8)))  float  v8f;

#define BB   8
#define NB   1024
#define NS   1024
#define CC   768
#define HH   12
#define DH   64
#define KSEL 256
#define C4   3072
#define LN_EPS 1e-5f

static __device__ __forceinline__ v8f wmma_bf16(v16bf a, v16bf b, v8f c) {
  // D = A(16x32 bf16) * B(32x16 bf16) + C(16x16 f32)
  return __builtin_amdgcn_wmma_f32_16x16x32_bf16(
      /*neg_a=*/false, a, /*neg_b=*/false, b,
      /*c_mod=*/(short)0, c, /*reuse_a=*/false, /*reuse_b=*/false);
}

// A-matrix 16x32 bf16 fragment (ISA 7.12.2): lane holds row (lane&15);
// halves 0..7 -> K = koff+0..7 ; halves 8..15 -> K = 16+koff+0..7,
// koff = (lane>=16) ? 8 : 0.   rowPtr points at this lane's row, K=0.
static __device__ __forceinline__ v16bf load_fragA(const bf16_t* rowPtr, int lane) {
  const int koff = (lane >> 4) << 3;
  v8bf lo = *(const v8bf*)(rowPtr + koff);
  v8bf hi = *(const v8bf*)(rowPtr + 16 + koff);
  v16bf a;
#pragma unroll
  for (int i = 0; i < 8; ++i) { a[i] = lo[i]; a[8 + i] = hi[i]; }
  return a;
}

// B-matrix 32x16 bf16 fragment: lane holds column (lane&15);
// halves 0..7 -> K = koff+0..7 ; halves 8..15 -> K = 16+koff+0..7,
// koff = (lane>=16) ? 8 : 0.  colPtr points at K=0 of this
// lane's column, K contiguous in memory.
static __device__ __forceinline__ v16bf load_fragB(const bf16_t* colPtr, int lane) {
  const bf16_t* p = colPtr + ((lane >> 4) << 3);
  v8bf lo = *(const v8bf*)p;
  v8bf hi = *(const v8bf*)(p + 16);
  v16bf b;
#pragma unroll
  for (int i = 0; i < 8; ++i) { b[i] = lo[i]; b[8 + i] = hi[i]; }
  return b;
}

// ---------------------------------------------------------------------------
// f32 -> bf16 convert
__global__ void cvt_bf16_kernel(const float* __restrict__ in,
                                bf16_t* __restrict__ out, int n) {
  int i = blockIdx.x * blockDim.x + threadIdx.x;
  if (i < n) out[i] = (bf16_t)in[i];
}

// exact GELU then convert to bf16
__global__ void gelu_cvt_kernel(const float* __restrict__ in,
                                bf16_t* __restrict__ out, int n) {
  int i = blockIdx.x * blockDim.x + threadIdx.x;
  if (i < n) {
    float x = in[i];
    float y = 0.5f * x * (1.0f + erff(x * 0.70710678118654752f));
    out[i] = (bf16_t)y;
  }
}

// V [B*NS, CC] -> Vt [B, H, DH, NS]  (makes attn@V B-fragments K-contiguous)
__global__ void transpose_v_kernel(const bf16_t* __restrict__ v,
                                   bf16_t* __restrict__ vt) {
  int i = blockIdx.x * blockDim.x + threadIdx.x;   // over B*NS*CC
  if (i >= BB * NS * CC) return;
  int c = i % CC;
  int m = (i / CC) % NS;
  int b = i / (CC * NS);
  int h = c / DH, d = c % DH;
  vt[(((size_t)b * HH + h) * DH + d) * NS + m] = v[i];
}

// ---------------------------------------------------------------------------
// Generic bf16 GEMM: out[M,N] = A[M,K] @ W[N,K]^T + bias.
// One wave per 32x32 output block (2x2 register tiles): per k-step,
// 2 A-frags + 2 B-frags feed 4 v_wmma -> loads/wmma halved vs 1x1.
__global__ void gemm_bf16_kernel(const bf16_t* __restrict__ A,
                                 const bf16_t* __restrict__ W,
                                 const float* __restrict__ bias,
                                 float* __restrict__ outF,
                                 bf16_t* __restrict__ outB,
                                 int M, int N, int K) {
  const int wave = (blockIdx.x * blockDim.x + threadIdx.x) >> 5;
  const int lane = threadIdx.x & 31;
  const int bTilesN = N >> 5;
  const int nBlocks = (M >> 5) * bTilesN;
  if (wave >= nBlocks) return;              // wave-uniform: EXEC stays all-1s
  const int bm = wave / bTilesN;
  const int bn = wave % bTilesN;
  const int r  = lane & 15;

  const bf16_t* Arow0 = A + (size_t)(bm * 32 + r) * K;
  const bf16_t* Arow1 = Arow0 + (size_t)16 * K;
  const bf16_t* Wrow0 = W + (size_t)(bn * 32 + r) * K;
  const bf16_t* Wrow1 = Wrow0 + (size_t)16 * K;

  v8f acc00 = {}, acc01 = {}, acc10 = {}, acc11 = {};
  for (int kk = 0; kk < K; kk += 32) {
    __builtin_prefetch(Arow0 + kk + 128, 0, 1);
    __builtin_prefetch(Arow1 + kk + 128, 0, 1);
    __builtin_prefetch(Wrow0 + kk + 128, 0, 1);
    __builtin_prefetch(Wrow1 + kk + 128, 0, 1);
    v16bf a0 = load_fragA(Arow0 + kk, lane);
    v16bf a1 = load_fragA(Arow1 + kk, lane);
    v16bf b0 = load_fragB(Wrow0 + kk, lane);
    v16bf b1 = load_fragB(Wrow1 + kk, lane);
    acc00 = wmma_bf16(a0, b0, acc00);
    acc01 = wmma_bf16(a0, b1, acc01);
    acc10 = wmma_bf16(a1, b0, acc10);
    acc11 = wmma_bf16(a1, b1, acc11);
  }

  const int c0 = bn * 32 + (lane & 15);
  const int c1 = c0 + 16;
  const float bv0 = bias ? bias[c0] : 0.0f;
  const float bv1 = bias ? bias[c1] : 0.0f;
  const int r0 = bm * 32 + ((lane >> 4) << 3);
  const int r1 = r0 + 16;
#pragma unroll
  for (int j = 0; j < 8; ++j) {
    float v00 = acc00[j] + bv0, v01 = acc01[j] + bv1;
    float v10 = acc10[j] + bv0, v11 = acc11[j] + bv1;
    size_t o00 = (size_t)(r0 + j) * N + c0, o01 = (size_t)(r0 + j) * N + c1;
    size_t o10 = (size_t)(r1 + j) * N + c0, o11 = (size_t)(r1 + j) * N + c1;
    if (outF) { outF[o00] = v00; outF[o01] = v01; outF[o10] = v10; outF[o11] = v11; }
    if (outB) { outB[o00] = (bf16_t)v00; outB[o01] = (bf16_t)v01;
                outB[o10] = (bf16_t)v10; outB[o11] = (bf16_t)v11; }
  }
}

// ---------------------------------------------------------------------------
// Pass 1: per-(b,h,32-row block) QK^T with rolling row-max in registers.
// 4 persistent A-frags (32 q-rows); each K-tile's 2 B-frags feed 4 wmma.
// rowmax[b,h,n] = max_m (q.k / sqrt(Dh)).  Logits never hit memory.
__global__ void saliency_rowmax_kernel(const bf16_t* __restrict__ Q,
                                       const bf16_t* __restrict__ Km,
                                       float* __restrict__ rowmax) {
  const int wave  = (blockIdx.x * blockDim.x + threadIdx.x) >> 5;
  const int lane  = threadIdx.x & 31;
  const int mtPer = NB / 32;                              // 32
  const int b  = wave / (HH * mtPer);
  const int h  = (wave / mtPer) % HH;
  const int mt = wave % mtPer;
  if (b >= BB) return;
  const int r = lane & 15;

  const bf16_t* qrow0 = Q + (size_t)(b * NB + mt * 32 + r) * CC + h * DH;
  const bf16_t* qrow1 = qrow0 + (size_t)16 * CC;
  v16bf a00 = load_fragA(qrow0, lane);       // rows 0..15, K 0..31
  v16bf a01 = load_fragA(qrow0 + 32, lane);  // rows 0..15, K 32..63
  v16bf a10 = load_fragA(qrow1, lane);       // rows 16..31, K 0..31
  v16bf a11 = load_fragA(qrow1 + 32, lane);  // rows 16..31, K 32..63

  float mx0[8], mx1[8];
#pragma unroll
  for (int j = 0; j < 8; ++j) { mx0[j] = -3.0e38f; mx1[j] = -3.0e38f; }

  for (int ct = 0; ct < NS / 16; ++ct) {
    const bf16_t* krow = Km + (size_t)(b * NS + ct * 16 + r) * CC + h * DH;
    v16bf b0 = load_fragB(krow, lane);
    v16bf b1 = load_fragB(krow + 32, lane);
    v8f acc0 = {}, acc1 = {};
    acc0 = wmma_bf16(a00, b0, acc0);
    acc0 = wmma_bf16(a01, b1, acc0);
    acc1 = wmma_bf16(a10, b0, acc1);
    acc1 = wmma_bf16(a11, b1, acc1);
#pragma unroll
    for (int j = 0; j < 8; ++j) {
      mx0[j] = fmaxf(mx0[j], acc0[j] * 0.125f);
      mx1[j] = fmaxf(mx1[j], acc1[j] * 0.125f);
    }
  }

  // reduce over the 16 lanes of each half (columns of the tile)
#pragma unroll
  for (int j = 0; j < 8; ++j) {
    float v0 = mx0[j], v1 = mx1[j];
    v0 = fmaxf(v0, __shfl_xor(v0, 1));  v1 = fmaxf(v1, __shfl_xor(v1, 1));
    v0 = fmaxf(v0, __shfl_xor(v0, 2));  v1 = fmaxf(v1, __shfl_xor(v1, 2));
    v0 = fmaxf(v0, __shfl_xor(v0, 4));  v1 = fmaxf(v1, __shfl_xor(v1, 4));
    v0 = fmaxf(v0, __shfl_xor(v0, 8));  v1 = fmaxf(v1, __shfl_xor(v1, 8));
    mx0[j] = v0; mx1[j] = v1;
  }
  if ((lane & 15) == 0) {
    const int rb = mt * 32 + ((lane >> 4) << 3);
    float* dst = rowmax + (size_t)(b * HH + h) * NB;
#pragma unroll
    for (int j = 0; j < 8; ++j) {
      dst[rb + j]      = mx0[j];
      dst[rb + 16 + j] = mx1[j];
    }
  }
}

// query_scores[b,n] = mean_h rowmax[b,h,n]   (deterministic, no atomics)
__global__ void mean_heads_kernel(const float* __restrict__ rowmax,
                                  float* __restrict__ qs) {
  int i = blockIdx.x * blockDim.x + threadIdx.x;
  if (i >= BB * NB) return;
  int b = i / NB, n = i % NB;
  float s = 0.0f;
#pragma unroll
  for (int h = 0; h < HH; ++h) s += rowmax[(size_t)(b * HH + h) * NB + n];
  qs[i] = s * (1.0f / HH);
}

// ---------------------------------------------------------------------------
// Deterministic stable top-k (descending, ties -> lower index), matching
// jax.lax.top_k.  One block per batch.
__global__ void topk_kernel(const float* __restrict__ qs,
                            int* __restrict__ idx_ws,
                            float* __restrict__ idx_out) {
  __shared__ float s[NB];
  const int b = blockIdx.x, t = threadIdx.x;
  s[t] = qs[b * NB + t];
  __syncthreads();
  const float mine = s[t];
  int rank = 0;
  for (int j = 0; j < NB; ++j) {
    float v = s[j];
    rank += (v > mine) || (v == mine && j < t);
  }
  if (rank < KSEL) {
    idx_ws[b * KSEL + rank]  = t;
    idx_out[b * KSEL + rank] = (float)t;
  }
}

// ---------------------------------------------------------------------------
// Pass 2: recompute logits only for the 256 selected rows, softmax in LDS,
// write selected_attn (f32 output) and fused attn@V (bf16 WMMA, Vt layout
// so B fragments are K-contiguous vector loads).
// One wave per (b, h, 16-row tile of the 256 selected rows).
// Dynamic LDS: 16*1024 f32 logits (64KB) + 16*1024 bf16 probs (32KB) = 96KB.
__global__ void sel_attn_kernel(const bf16_t* __restrict__ Q,
                                const bf16_t* __restrict__ Km,
                                const bf16_t* __restrict__ Vt,
                                const int* __restrict__ topk,
                                float* __restrict__ attn_out,
                                bf16_t* __restrict__ attnV) {
  extern __shared__ char smem[];
  float*  lg = (float*)smem;                                      // [16][NS]
  bf16_t* pb = (bf16_t*)(smem + (size_t)16 * NS * sizeof(float)); // [16][NS]

  const int lane = threadIdx.x & 31;
  const int wid  = blockIdx.x;                 // b*HH*16 + h*16 + mt
  const int mt   = wid & 15;
  const int h    = (wid >> 4) % HH;
  const int b    = wid / (HH * 16);
  const int r    = lane & 15;

  const int srow = topk[b * KSEL + mt * 16 + r];
  const bf16_t* qrow = Q + (size_t)(b * NB + srow) * CC + h * DH;
  v16bf a0 = load_fragA(qrow, lane);
  v16bf a1 = load_fragA(qrow + 32, lane);

  for (int ct = 0; ct < NS / 16; ++ct) {
    const bf16_t* krow = Km + (size_t)(b * NS + ct * 16 + r) * CC + h * DH;
    v16bf b0 = load_fragB(krow, lane);
    v16bf b1 = load_fragB(krow + 32, lane);
    v8f acc = {};
    acc = wmma_bf16(a0, b0, acc);
    acc = wmma_bf16(a1, b1, acc);
    const int rb = (lane >> 4) << 3;
#pragma unroll
    for (int j = 0; j < 8; ++j)
      lg[(rb + j) * NS + ct * 16 + (lane & 15)] = acc[j] * 0.125f;
  }
  __syncthreads();

  // per-row softmax: lane i handles row i (i < 16)
  if (lane < 16) {
    float m = -3.0e38f;
    for (int j = 0; j < NS; ++j) m = fmaxf(m, lg[lane * NS + j]);
    float sum = 0.0f;
    for (int j = 0; j < NS; ++j) {
      float e = __expf(lg[lane * NS + j] - m);
      lg[lane * NS + j] = e;
      sum += e;
    }
    const float inv = 1.0f / sum;
    float* outRow =
        attn_out + ((size_t)((b * HH + h) * KSEL) + mt * 16 + lane) * NS;
    for (int j = 0; j < NS; ++j) {
      float p = lg[lane * NS + j] * inv;
      outRow[j] = p;
      pb[lane * NS + j] = (bf16_t)p;
    }
  }
  __syncthreads();

  // attn @ V : A from LDS probs, B from Vt[b][h][d][m] (K=m contiguous)
  const bf16_t* vt_base = Vt + (((size_t)b * HH + h) * DH) * NS;
  v8f acc4[4] = {};
  for (int kk = 0; kk < NS; kk += 32) {
    v16bf a = load_fragA(pb + (size_t)r * NS + kk, lane);
#pragma unroll
    for (int n0 = 0; n0 < 4; ++n0) {
      const bf16_t* vcol = vt_base + (size_t)(n0 * 16 + (lane & 15)) * NS + kk;
      v16bf bb = load_fragB(vcol, lane);
      acc4[n0] = wmma_bf16(a, bb, acc4[n0]);
    }
  }
  const int rb = (lane >> 4) << 3;
#pragma unroll
  for (int n0 = 0; n0 < 4; ++n0)
#pragma unroll
    for (int j = 0; j < 8; ++j)
      attnV[(size_t)(b * KSEL + mt * 16 + rb + j) * CC + h * DH + n0 * 16 +
            (lane & 15)] = (bf16_t)acc4[n0][j];
}

// ---------------------------------------------------------------------------
// x = LN(base[topk] + selected_out); writes f32 (kept for residual) + bf16
__global__ void gather_add_ln_kernel(const float* __restrict__ base,
                                     const float* __restrict__ selOut,
                                     const int* __restrict__ topk,
                                     const float* __restrict__ g,
                                     const float* __restrict__ be,
                                     float* __restrict__ xf,
                                     bf16_t* __restrict__ xb) {
  __shared__ float red[256];
  __shared__ float buf[CC];
  const int row = blockIdx.x;                 // b*KSEL + k
  const int b = row / KSEL, k = row % KSEL;
  const int src = topk[b * KSEL + k];
  const float* p0 = base + (size_t)(b * NB + src) * CC;
  const float* p1 = selOut + (size_t)row * CC;
  const int t = threadIdx.x;

  float s = 0.0f, s2 = 0.0f;
  for (int c = t; c < CC; c += 256) {
    float v = p0[c] + p1[c];
    buf[c] = v; s += v; s2 += v * v;
  }
  red[t] = s; __syncthreads();
  for (int st = 128; st > 0; st >>= 1) { if (t < st) red[t] += red[t + st]; __syncthreads(); }
  const float mu = red[0] / CC;
  __syncthreads();
  red[t] = s2; __syncthreads();
  for (int st = 128; st > 0; st >>= 1) { if (t < st) red[t] += red[t + st]; __syncthreads(); }
  const float var = red[0] / CC - mu * mu;
  const float inv = rsqrtf(var + LN_EPS);
  for (int c = t; c < CC; c += 256) {
    float y = (buf[c] - mu) * inv * g[c] + be[c];
    xf[(size_t)row * CC + c] = y;
    xb[(size_t)row * CC + c] = (bf16_t)y;
  }
}

// out = LN(x + h)
__global__ void add_ln_kernel(const float* __restrict__ xf,
                              const float* __restrict__ hf,
                              const float* __restrict__ g,
                              const float* __restrict__ be,
                              float* __restrict__ out) {
  __shared__ float red[256];
  __shared__ float buf[CC];
  const int row = blockIdx.x;
  const int t = threadIdx.x;
  const float* p0 = xf + (size_t)row * CC;
  const float* p1 = hf + (size_t)row * CC;

  float s = 0.0f, s2 = 0.0f;
  for (int c = t; c < CC; c += 256) {
    float v = p0[c] + p1[c];
    buf[c] = v; s += v; s2 += v * v;
  }
  red[t] = s; __syncthreads();
  for (int st = 128; st > 0; st >>= 1) { if (t < st) red[t] += red[t + st]; __syncthreads(); }
  const float mu = red[0] / CC;
  __syncthreads();
  red[t] = s2; __syncthreads();
  for (int st = 128; st > 0; st >>= 1) { if (t < st) red[t] += red[t + st]; __syncthreads(); }
  const float var = red[0] / CC - mu * mu;
  const float inv = rsqrtf(var + LN_EPS);
  for (int c = t; c < CC; c += 256)
    out[(size_t)row * CC + c] = (buf[c] - mu) * inv * g[c] + be[c];
}

// ---------------------------------------------------------------------------
extern "C" void kernel_launch(void* const* d_in, const int* in_sizes, int n_in,
                              void* d_out, int out_size, void* d_ws, size_t ws_size,
                              hipStream_t stream) {
  (void)in_sizes; (void)n_in; (void)out_size; (void)ws_size;

  const float* base = (const float*)d_in[0];
  const float* samp = (const float*)d_in[1];
  const float* Wq = (const float*)d_in[2];  const float* bq = (const float*)d_in[3];
  const float* Wk = (const float*)d_in[4];  const float* bk = (const float*)d_in[5];
  const float* Wv = (const float*)d_in[6];  const float* bv = (const float*)d_in[7];
  const float* Wo = (const float*)d_in[8];  const float* bo = (const float*)d_in[9];
  const float* g1 = (const float*)d_in[10]; const float* b1 = (const float*)d_in[11];
  const float* g2 = (const float*)d_in[12]; const float* b2 = (const float*)d_in[13];
  const float* Wfc1 = (const float*)d_in[14]; const float* bfc1 = (const float*)d_in[15];
  const float* Wfc2 = (const float*)d_in[16]; const float* bfc2 = (const float*)d_in[17];

  // ---- output layout (flat f32, tuple order) ----
  float* out_x    = (float*)d_out;                            // [8,256,768]
  float* out_qs   = out_x + (size_t)BB * KSEL * CC;           // [8,1024]
  float* out_attn = out_qs + (size_t)BB * NB;                 // [8,12,256,1024]
  float* out_idx  = out_attn + (size_t)BB * HH * KSEL * NS;   // [8,256]

  // ---- workspace layout ----
  char* w = (char*)d_ws;
  size_t off = 0;
  auto take = [&](size_t bytes) -> char* {
    char* p = w + off;
    off += (bytes + 255) & ~(size_t)255;
    return p;
  };
  bf16_t* base_bf = (bf16_t*)take((size_t)BB * NB * CC * 2);
  bf16_t* samp_bf = (bf16_t*)take((size_t)BB * NS * CC * 2);
  bf16_t* wq_bf   = (bf16_t*)take((size_t)CC * CC * 2);
  bf16_t* wk_bf   = (bf16_t*)take((size_t)CC * CC * 2);
  bf16_t* wv_bf   = (bf16_t*)take((size_t)CC * CC * 2);
  bf16_t* wo_bf   = (bf16_t*)take((size_t)CC * CC * 2);
  bf16_t* wfc1_bf = (bf16_t*)take((size_t)C4 * CC * 2);
  bf16_t* wfc2_bf = (bf16_t*)take((size_t)CC * C4 * 2);
  bf16_t* q_bf    = (bf16_t*)take((size_t)BB * NB * CC * 2);
  bf16_t* k_bf    = (bf16_t*)take((size_t)BB * NS * CC * 2);
  bf16_t* v_bf    = (bf16_t*)take((size_t)BB * NS * CC * 2);
  bf16_t* vt_bf   = (bf16_t*)take((size_t)BB * HH * DH * NS * 2);
  float*  rowmax  = (float*) take((size_t)BB * HH * NB * 4);
  int*    topk_ws = (int*)   take((size_t)BB * KSEL * 4);
  bf16_t* attnV   = (bf16_t*)take((size_t)BB * KSEL * CC * 2);
  float*  selOut  = (float*) take((size_t)BB * KSEL * CC * 4);
  float*  x_f     = (float*) take((size_t)BB * KSEL * CC * 4);
  bf16_t* x_bf    = (bf16_t*)take((size_t)BB * KSEL * CC * 2);
  float*  h1_f    = (float*) take((size_t)BB * KSEL * C4 * 4);
  bf16_t* h1_bf   = (bf16_t*)take((size_t)BB * KSEL * C4 * 2);
  float*  h2_f    = (float*) take((size_t)BB * KSEL * CC * 4);

  // ---- 1. convert inputs/weights to bf16 ----
  const int nTok = BB * NB * CC;        // 6291456
  cvt_bf16_kernel<<<(nTok + 255) / 256, 256, 0, stream>>>(base, base_bf, nTok);
  cvt_bf16_kernel<<<(nTok + 255) / 256, 256, 0, stream>>>(samp, samp_bf, nTok);
  const int nW = CC * CC;
  cvt_bf16_kernel<<<(nW + 255) / 256, 256, 0, stream>>>(Wq, wq_bf, nW);
  cvt_bf16_kernel<<<(nW + 255) / 256, 256, 0, stream>>>(Wk, wk_bf, nW);
  cvt_bf16_kernel<<<(nW + 255) / 256, 256, 0, stream>>>(Wv, wv_bf, nW);
  cvt_bf16_kernel<<<(nW + 255) / 256, 256, 0, stream>>>(Wo, wo_bf, nW);
  const int nW4 = C4 * CC;
  cvt_bf16_kernel<<<(nW4 + 255) / 256, 256, 0, stream>>>(Wfc1, wfc1_bf, nW4);
  cvt_bf16_kernel<<<(nW4 + 255) / 256, 256, 0, stream>>>(Wfc2, wfc2_bf, nW4);

  // ---- 2. Q/K/V projections (bf16 WMMA, 2x2 blocked, bf16 out) ----
  {
    const int M = BB * NB, N = CC, K = CC;
    const int blocks32 = (M / 32) * (N / 32);       // 6144 waves
    const int grid = (blocks32 + 7) / 8;            // 8 waves / workgroup
    gemm_bf16_kernel<<<grid, 256, 0, stream>>>(base_bf, wq_bf, bq, nullptr, q_bf, M, N, K);
    gemm_bf16_kernel<<<grid, 256, 0, stream>>>(samp_bf, wk_bf, bk, nullptr, k_bf, M, N, K);
    gemm_bf16_kernel<<<grid, 256, 0, stream>>>(samp_bf, wv_bf, bv, nullptr, v_bf, M, N, K);
  }
  transpose_v_kernel<<<(nTok + 255) / 256, 256, 0, stream>>>(v_bf, vt_bf);

  // ---- 3. fused QK^T row-max (no logits materialization) ----
  {
    const int waves = BB * HH * (NB / 32);          // 3072
    saliency_rowmax_kernel<<<waves / 4, 128, 0, stream>>>(q_bf, k_bf, rowmax);
  }
  mean_heads_kernel<<<(BB * NB + 255) / 256, 256, 0, stream>>>(rowmax, out_qs);

  // ---- 4. deterministic stable top-k ----
  topk_kernel<<<BB, NB, 0, stream>>>(out_qs, topk_ws, out_idx);

  // ---- 5. selected attention: logits -> softmax -> attn@V (fused) ----
  {
    const int blocks = BB * HH * (KSEL / 16);       // 1536 single-wave WGs
    const size_t lds = (size_t)16 * NS * sizeof(float) + (size_t)16 * NS * 2;  // 96KB
    sel_attn_kernel<<<blocks, 32, lds, stream>>>(q_bf, k_bf, vt_bf, topk_ws,
                                                 out_attn, attnV);
  }

  // ---- 6. output projection Wo ----
  {
    const int M = BB * KSEL, N = CC, K = CC;
    const int blocks32 = (M / 32) * (N / 32);       // 1536
    gemm_bf16_kernel<<<(blocks32 + 7) / 8, 256, 0, stream>>>(attnV, wo_bf, bo,
                                                             selOut, nullptr, M, N, K);
  }

  // ---- 7. residual + LN1 ----
  gather_add_ln_kernel<<<BB * KSEL, 256, 0, stream>>>(base, selOut, topk_ws,
                                                      g1, b1, x_f, x_bf);

  // ---- 8. MLP: fc1 -> exact GELU -> fc2 ----
  {
    const int M = BB * KSEL, N = C4, K = CC;
    const int blocks32 = (M / 32) * (N / 32);       // 6144
    gemm_bf16_kernel<<<(blocks32 + 7) / 8, 256, 0, stream>>>(x_bf, wfc1_bf, bfc1,
                                                             h1_f, nullptr, M, N, K);
  }
  {
    const int n = BB * KSEL * C4;
    gelu_cvt_kernel<<<(n + 255) / 256, 256, 0, stream>>>(h1_f, h1_bf, n);
  }
  {
    const int M = BB * KSEL, N = CC, K = C4;
    const int blocks32 = (M / 32) * (N / 32);       // 1536
    gemm_bf16_kernel<<<(blocks32 + 7) / 8, 256, 0, stream>>>(h1_bf, wfc2_bf, bfc2,
                                                             h2_f, nullptr, M, N, K);
  }

  // ---- 9. residual + LN2 -> final x ----
  add_ln_kernel<<<BB * KSEL, 256, 0, stream>>>(x_f, h2_f, g2, b2, out_x);
}
